// gen_C_20272245637610
// MI455X (gfx1250) — matrix operand order repaired and checked by reading, not compiled, not run on hardware
//
#include <hip/hip_runtime.h>
#include <hip/hip_bf16.h>

// MI455X / gfx1250 implementation.
//
// Reference: per-edge MLP 3 -> 512 -> (512x512)x3 -> 4, then COO scatter-add
// into a 12000x12000 dense fp32 matrix.
//   E = 192,000 rows; hidden layers = batched [E,512]x[512,512] GEMM
//   (~302 GFLOP) -> f16 WMMA with fp32 accumulate.
//   Output = 576 MB fp32 -> memset + float atomics (~25 us of HBM traffic).
//
// Round-1 change: 64-row M-tiles per block (was 16) to quadruple B-matrix
// reuse out of L2: per-block compute/B-bytes goes 16 -> 64 FLOP/byte, cutting
// total L2 weight traffic from ~19 GB to ~4.7 GB across the 3 layers, which
// was the dominant term vs ~60 us of WMMA math.
//
// Workspace layout (d_ws):
//   [0]                f16 packed hidden weights: 3 * 512*512 halves (1.5 MB)
//   [aligned]          actA: E*512 f16 (196.6 MB)
//   [next]             actB: E*512 f16 (196.6 MB)

typedef __attribute__((ext_vector_type(16))) _Float16 v16h;
typedef __attribute__((ext_vector_type(8)))  float    v8f;

#define NODES 512
#define KTILES 16   // 512 / 32
#define NTILES 32   // 512 / 16
#define BM     64   // rows per block in hidden GEMM

// ---------------------------------------------------------------------------
// Pack one hidden layer's weights W[512,512] (row-major fp32, row=k, col=n)
// into per-lane B fragments for v_wmma_f32_16x16x32_f16.
// B is 32x16 (KxN). Lane layout (contiguous k order, same as the A fragments):
//   lanes 0..15  : column n = lane,    K = kt*32 + 0..15   (16 halves)
//   lanes 16..31 : column n = lane-16, K = kt*32 + 16..31  (16 halves)
// Fragment storage: Wp[((l*KTILES + kt)*NTILES + nt)*32 + lane][16 halves].
// ---------------------------------------------------------------------------
__global__ void k_pack_w(const float* __restrict__ Wh, _Float16* __restrict__ Wp) {
    int t    = blockIdx.x * blockDim.x + threadIdx.x;   // 3*16*32*32 threads
    int lane = t & 31;
    int tile = t >> 5;
    int nt   = tile & 31;
    int kt   = (tile >> 5) & 15;
    int l    = tile >> 9;

    int n  = nt * 16 + (lane & 15);
    int kb = kt * 32 + (lane >> 4) * 16;

    const float* W = Wh + (long long)l * NODES * NODES;
    long long di = (((long long)(l * (KTILES * NTILES) + kt * NTILES + nt)) * 32 + lane) * 16;
#pragma unroll
    for (int j = 0; j < 16; ++j)
        Wp[di + j] = (_Float16)W[(long long)(kb + j) * NODES + n];
}

// ---------------------------------------------------------------------------
// Input layer: h0[e][n] = relu(x[e][0..2] . W_in[0..2][n] + b_in[n]), f16 out.
// ---------------------------------------------------------------------------
__global__ void k_input(const float* __restrict__ x, const float* __restrict__ Win,
                        const float* __restrict__ bin, _Float16* __restrict__ out,
                        long long total) {
    long long i = (long long)blockIdx.x * blockDim.x + threadIdx.x;
    if (i >= total) return;
    long long e = i >> 9;
    int n = (int)(i & (NODES - 1));
    const float* xr = x + e * 3;
    float v = xr[0] * Win[n] + xr[1] * Win[NODES + n] + xr[2] * Win[2 * NODES + n] + bin[n];
    out[i] = (_Float16)(v > 0.f ? v : 0.f);
}

// ---------------------------------------------------------------------------
// Hidden layer GEMM: actOut = relu(actIn[64 rows/block, 512] x W[512,512] + b)
// 8 waves/block; wave w owns columns [w*64, w*64+64) across all 64 rows:
// 4 M-tiles x 4 N-tiles = 16 WMMA accumulators per wave; B fragments loaded
// once per K-step and reused across the 4 A fragments.
// A tile (64x512 f16, 64 KB) staged in LDS once per block.
// E must be a multiple of 64 (it is: 192,000) -> no guards, EXEC all-ones.
// ---------------------------------------------------------------------------
__global__ __launch_bounds__(256) void k_hidden(const _Float16* __restrict__ actIn,
                                                const _Float16* __restrict__ Wp,
                                                const float* __restrict__ bias,
                                                _Float16* __restrict__ actOut) {
    __shared__ _Float16 At[BM * NODES];          // 64 KB
    const int tid  = threadIdx.x;
    const int wave = tid >> 5;
    const int lane = tid & 31;
    const long long rowBase = (long long)blockIdx.x * BM;

    // Cooperative copy: the 64-row stripe is contiguous in memory (64 KB).
    {
        const uint4* src = (const uint4*)(actIn + rowBase * NODES);
        uint4* dst = (uint4*)At;
#pragma unroll
        for (int i = 0; i < 16; ++i)
            dst[tid + i * 256] = src[tid + i * 256];
    }
    __syncthreads();

    v8f acc[4][4] = {};                          // [mt][nt-within-wave]

    // 16-bit A 16x32 fragment layout (matches packed B):
    //   lanes 0..15 : row m=lane,    K = 0..15
    //   lanes 16..31: row m=lane-16, K = 16..31
    const int rowA = lane & 15;
    const int koff = (lane >> 4) * 16;
    const uint4* lp = (const uint4*)At;          // 8 halves per uint4
    const v16h* Bp = (const v16h*)Wp;

    for (int kt = 0; kt < KTILES; ++kt) {
        v16h b[4];
#pragma unroll
        for (int j = 0; j < 4; ++j)
            b[j] = Bp[(kt * NTILES + wave * 4 + j) * 32 + lane];
#pragma unroll
        for (int mt = 0; mt < 4; ++mt) {
            v16h a;
            int hbase = (mt * 16 + rowA) * NODES + kt * 32 + koff;  // mult of 8
            ((uint4*)&a)[0] = lp[hbase >> 3];
            ((uint4*)&a)[1] = lp[(hbase + 8) >> 3];
#pragma unroll
            for (int j = 0; j < 4; ++j)
                acc[mt][j] = __builtin_amdgcn_wmma_f32_16x16x32_f16(
                    false, a, false, b[j], (short)0, acc[mt][j], false, false);
        }
    }

    // C/D layout: VGPR r -> (M=r, N=lane) lanes 0..15; (M=8+r, N=lane-16) hi.
    const int mrow = (lane >> 4) * 8;
    const int ncol = lane & 15;
#pragma unroll
    for (int mt = 0; mt < 4; ++mt) {
#pragma unroll
        for (int j = 0; j < 4; ++j) {
            int n = (wave * 4 + j) * 16 + ncol;
            float bv = bias[n];
#pragma unroll
            for (int r = 0; r < 8; ++r) {
                float v = acc[mt][j][r] + bv;
                v = v > 0.f ? v : 0.f;
                actOut[(rowBase + mt * 16 + mrow + r) * NODES + n] = (_Float16)v;
            }
        }
    }
}

// ---------------------------------------------------------------------------
// Output layer + fused scatter: one wave per edge.
// vals[e,o] = h[e,:] . W_out[:,o] + b_out[o], o = mi*2+mj;
// dense[(2r+mi)*dim + 2c+mj] += vals (float atomics; dups only at boundaries).
// ---------------------------------------------------------------------------
__global__ __launch_bounds__(256) void k_out(const _Float16* __restrict__ act,
                                             const float* __restrict__ Wout,
                                             const float* __restrict__ bout,
                                             const int* __restrict__ rows,
                                             const int* __restrict__ cols,
                                             float* __restrict__ dout,
                                             int E, int dim) {
    const int wave = threadIdx.x >> 5;
    const int lane = threadIdx.x & 31;
    const int e = blockIdx.x * 8 + wave;
    if (e >= E) return;

    const _Float16* h = act + (long long)e * NODES;
    float s0 = 0.f, s1 = 0.f, s2 = 0.f, s3 = 0.f;
    for (int k = lane; k < NODES; k += 32) {
        float hv = (float)h[k];
        float4 w = ((const float4*)Wout)[k];     // W_out is [512,4] row-major
        s0 += hv * w.x; s1 += hv * w.y; s2 += hv * w.z; s3 += hv * w.w;
    }
#pragma unroll
    for (int off = 16; off > 0; off >>= 1) {
        s0 += __shfl_xor(s0, off);
        s1 += __shfl_xor(s1, off);
        s2 += __shfl_xor(s2, off);
        s3 += __shfl_xor(s3, off);
    }
    if (lane == 0) {
        long long r = (long long)rows[e] * 2;
        long long c = (long long)cols[e] * 2;
        atomicAdd(&dout[r * dim + c],           s0 + bout[0]);
        atomicAdd(&dout[r * dim + c + 1],       s1 + bout[1]);
        atomicAdd(&dout[(r + 1) * dim + c],     s2 + bout[2]);
        atomicAdd(&dout[(r + 1) * dim + c + 1], s3 + bout[3]);
    }
}

extern "C" void kernel_launch(void* const* d_in, const int* in_sizes, int n_in,
                              void* d_out, int out_size, void* d_ws, size_t ws_size,
                              hipStream_t stream) {
    (void)n_in; (void)ws_size;
    const float* CK   = (const float*)d_in[0];
    const float* Win  = (const float*)d_in[1];
    const float* bin  = (const float*)d_in[2];
    const float* Wh   = (const float*)d_in[3];
    const float* bh   = (const float*)d_in[4];
    const float* Wout = (const float*)d_in[5];
    const float* bout = (const float*)d_in[6];
    const int*   rows = (const int*)d_in[7];
    const int*   cols = (const int*)d_in[8];
    float* dout = (float*)d_out;

    const int E   = in_sizes[0] / 3;   // 192,000 (multiple of 64)
    const int Nn  = E / 32;            // N = 6000 (2*(KNN+1) = 32)
    const int dim = 2 * Nn;            // 12000

    // Workspace carve-up.
    char* p = (char*)d_ws;
    _Float16* Wpk = (_Float16*)p;
    p += ((size_t)3 * NODES * NODES * sizeof(_Float16) + 255) & ~(size_t)255;
    _Float16* actA = (_Float16*)p;
    p += (size_t)E * NODES * sizeof(_Float16);
    _Float16* actB = (_Float16*)p;

    // Zero the 576 MB dense output (capture-legal async memset).
    hipMemsetAsync(d_out, 0, (size_t)out_size * sizeof(float), stream);

    // Pack hidden weights to f16 B-fragment layout.
    k_pack_w<<<(3 * KTILES * NTILES * 32) / 256, 256, 0, stream>>>(Wh, Wpk);

    // Input layer.
    long long tot = (long long)E * NODES;
    k_input<<<(int)((tot + 255) / 256), 256, 0, stream>>>(CK, Win, bin, actA, tot);

    // Three hidden layers (ping-pong).
    k_hidden<<<E / BM, 256, 0, stream>>>(actA, Wpk,                     bh,             actB);
    k_hidden<<<E / BM, 256, 0, stream>>>(actB, Wpk + 1 * NODES * NODES, bh + 1 * NODES, actA);
    k_hidden<<<E / BM, 256, 0, stream>>>(actA, Wpk + 2 * NODES * NODES, bh + 2 * NODES, actB);

    // Output layer + scatter-add.
    k_out<<<(E + 7) / 8, 256, 0, stream>>>(actB, Wout, bout, rows, cols, dout, E, dim);
}
